// LocalCorrBlock_43344809951734
// MI455X (gfx1250) — matrix operand order repaired and checked by reading, not compiled, not run on hardware
//
#include <hip/hip_runtime.h>
#include <hip/hip_bf16.h>

typedef __attribute__((ext_vector_type(16))) _Float16 v16h;
typedef __attribute__((ext_vector_type(8)))  _Float16 v8h;
typedef __attribute__((ext_vector_type(8)))  float    v8f;

#define C_CH 256
#define H_0  64
#define W_0  96
#define NPIX (H_0 * W_0)

// Padded pyramid geometry (4 left/top pad, generous right/bottom slack so all
// 16-wide corner-tile reads stay in-bounds; zeros reproduce padding_mode='zeros').
//   L0: 74 x 116, pos offset     0
//   L1: 42 x  68, pos offset  8584
//   L2: 26 x  44, pos offset 11440
//   L3: 18 x  32, pos offset 12584
//   total 13160 positions x 256 ch (f16)

__global__ void zero_u32_kernel(unsigned int* p, int n) {
  int i = blockIdx.x * blockDim.x + threadIdx.x;
  int stride = gridDim.x * blockDim.x;
  for (; i < n; i += stride) p[i] = 0u;
}

// fmap0 [C,H,W] f32 -> f0h [H*W, C] f16  (channel-contiguous per pixel = GEMM A rows)
__global__ void conv_f0_kernel(const float* __restrict__ f0, _Float16* __restrict__ f0h) {
  int p = blockIdx.x, c = threadIdx.x;
  f0h[(size_t)p * C_CH + c] = (_Float16)f0[(size_t)c * NPIX + p];
}

// fmap1 [C,H,W] f32 -> padded level-0 [pos, C] f16
__global__ void conv_f1_kernel(const float* __restrict__ f1, _Float16* __restrict__ f1p) {
  int p = blockIdx.x, c = threadIdx.x;
  int h = p / W_0, w = p % W_0;
  f1p[(((size_t)(h + 4)) * 116 + (w + 4)) * C_CH + c] = (_Float16)f1[(size_t)c * NPIX + p];
}

// 2x2 avg pool between padded levels (reads parent interior only)
__global__ void pool_kernel(const _Float16* __restrict__ src, _Float16* __restrict__ dst,
                            int Wi, int WpC, int WpP) {
  int pos = blockIdx.x, c = threadIdx.x;
  int by = pos / Wi, bx = pos % Wi;
  size_t p00 = (((size_t)(2 * by + 4)) * WpP + (2 * bx + 4)) * C_CH + c;
  size_t rs  = (size_t)WpP * C_CH;
  float v = (float)src[p00] + (float)src[p00 + C_CH] +
            (float)src[p00 + rs] + (float)src[p00 + rs + C_CH];
  dst[(((size_t)(by + 4)) * WpC + (bx + 4)) * C_CH + c] = (_Float16)(0.25f * v);
}

union AFrag { v16h v; v8h h[2]; };
union CFrag { v8f v; float f[8]; };

// One wave = one 16-pixel tile. D = f0_tile(16xK) x corners(Kx16) via
// v_wmma_f32_16x16x32_f16, staged in LDS, then 81-tap bilinear blend/scatter.
__global__ __launch_bounds__(32) void corr_wmma_kernel(const _Float16* __restrict__ f0h,
                                                       const _Float16* __restrict__ f1p,
                                                       float* __restrict__ out) {
  __shared__ float Dld[10][16][32];

  const int lvl  = blockIdx.y;
  const int tile = blockIdx.x;
  const int lane = threadIdx.x;
  const int g    = lane >> 4;   // K-half select per ISA 16-bit fragment layout
  const int ln   = lane & 15;

  int Wp, off, cols, h0, w0;
  if (lvl == 0)      { Wp = 116; off = 0;     cols = 16; h0 = tile / 6;        w0 = (tile % 6) * 16; }
  else if (lvl == 1) { Wp = 68;  off = 8584;  cols = 8;  h0 = (tile / 12) * 2; w0 = (tile % 12) * 8; }
  else if (lvl == 2) { Wp = 44;  off = 11440; cols = 4;  h0 = (tile / 24) * 4; w0 = (tile % 24) * 4; }
  else               { Wp = 32;  off = 12584; cols = 8;  h0 = (tile / 12) * 2; w0 = (tile % 12) * 8; }
  const int s   = 1 << lvl;
  const int by0 = h0 >> lvl;
  const int bx0 = w0 >> lvl;

  // ---- A fragments: 16 pixels (M) x 256 ch (K), kept in VGPRs for whole tile.
  // A(16x32,f16): lanes 0-15 hold K = kk*32 + {0..15}; lanes 16-31 +16 (matches B).
  const int hA = h0 + ln / cols;
  const int wA = w0 + ln % cols;
  const _Float16* ap = f0h + ((size_t)hA * W_0 + wA) * C_CH;
  AFrag a[8];
#pragma unroll
  for (int kk = 0; kk < 8; ++kk) {
    a[kk].h[0] = *(const v8h*)(ap + kk * 32 + 16 * g);
    a[kk].h[1] = *(const v8h*)(ap + kk * 32 + 16 * g + 8);
  }

  // ---- Corner dot-product tiles: D[t][m][j], t = corner row qy-(by0-4)
  const _Float16* lp = f1p + (size_t)off * C_CH;
  const int NT = (lvl == 0) ? 2 : 1;  // level 0 needs 25 corner columns -> 2 N-tiles
  for (int t = 0; t < 10; ++t) {
    const int rowIdx = by0 + t;                 // padded row of qy = by0-4+t
    for (int nt = 0; nt < NT; ++nt) {
      const int colIdx = bx0 + 16 * nt + ln;    // padded col of qx = bx0-4+16nt+ln
      // B(32x16,f16): lane n=L%16, K = kk*32 + 16g + {0..15} contiguous (same order as A).
      const _Float16* bp = lp + (((size_t)rowIdx) * Wp + colIdx) * C_CH + 16 * g;
      v8f acc = {};
#pragma unroll
      for (int kk = 0; kk < 8; ++kk) {
        v16h b = *(const v16h*)(bp + kk * 32);
        acc = __builtin_amdgcn_wmma_f32_16x16x32_f16(false, a[kk].v, false, b,
                                                     (short)0, acc, false, false);
      }
      CFrag cf; cf.v = acc;
#pragma unroll
      for (int r = 0; r < 8; ++r)               // C/D: lane col = L%16, rows 8g+r
        Dld[t][8 * g + r][16 * nt + ln] = cf.f[r];
    }
  }
  __syncthreads();

  // ---- 81-tap bilinear blend of corner dots, scaled by 1/sqrt(C)
  const float scale = 0.0625f;
  const float inv_s = 1.0f / (float)s;
  for (int idx = lane; idx < 16 * 81; idx += 32) {
    const int m   = idx / 81;
    const int tap = idx - m * 81;
    const int dyi = tap / 9;
    const int dxi = tap - dyi * 9;
    const int h = h0 + m / cols;
    const int w = w0 + m % cols;
    const float fx = (float)(w & (s - 1)) * inv_s;
    const float fy = (float)(h & (s - 1)) * inv_s;
    const int jx = dxi + (w >> lvl) - bx0;
    const int jy = dyi + (h >> lvl) - by0;
    const float d00 = Dld[jy    ][m][jx    ];
    const float d01 = Dld[jy    ][m][jx + 1];
    const float d10 = Dld[jy + 1][m][jx    ];
    const float d11 = Dld[jy + 1][m][jx + 1];
    const float v = (1.0f - fy) * ((1.0f - fx) * d00 + fx * d01) +
                    fy * ((1.0f - fx) * d10 + fx * d11);
    out[(((size_t)(lvl * 81 + tap)) * H_0 + h) * W_0 + w] = v * scale;
  }
}

extern "C" void kernel_launch(void* const* d_in, const int* in_sizes, int n_in,
                              void* d_out, int out_size, void* d_ws, size_t ws_size,
                              hipStream_t stream) {
  const float* f0 = (const float*)d_in[0];
  const float* f1 = (const float*)d_in[1];
  float* out = (float*)d_out;

  // Workspace: f0h (3.15 MB) then padded f16 pyramid (6.74 MB) -> ~9.9 MB total.
  _Float16* f0h = (_Float16*)d_ws;
  _Float16* f1p = f0h + (size_t)NPIX * C_CH;

  zero_u32_kernel<<<1024, 256, 0, stream>>>((unsigned int*)f1p, (13160 * C_CH) / 2);
  conv_f0_kernel<<<NPIX, 256, 0, stream>>>(f0, f0h);
  conv_f1_kernel<<<NPIX, 256, 0, stream>>>(f1, f1p);
  pool_kernel<<<32 * 48, 256, 0, stream>>>(f1p,
                                           f1p + (size_t)8584 * C_CH, 48, 68, 116);
  pool_kernel<<<16 * 24, 256, 0, stream>>>(f1p + (size_t)8584 * C_CH,
                                           f1p + (size_t)11440 * C_CH, 24, 44, 68);
  pool_kernel<<<8 * 12, 256, 0, stream>>>(f1p + (size_t)11440 * C_CH,
                                          f1p + (size_t)12584 * C_CH, 12, 32, 44);
  corr_wmma_kernel<<<dim3(384, 4), 32, 0, stream>>>(f0h, f1p, out);
}
